// MerelModel_91199335563338
// MI455X (gfx1250) — matrix operand order repaired and checked by reading, not compiled, not run on hardware
//
#include <hip/hip_runtime.h>
#include <hip/hip_bf16.h>

typedef __attribute__((ext_vector_type(16))) __bf16 v16bf;
typedef __attribute__((ext_vector_type(8)))  __bf16 v8bf;
typedef __attribute__((ext_vector_type(8)))  float  v8f;

// ---------------------------------------------------------------------------
// Pack f32 weight [N, Kw] row-major into zero-padded bf16 [Npad, Kpad].
// ---------------------------------------------------------------------------
__global__ void pack_w_bf16(const float* __restrict__ W, int N, int Kw,
                            __bf16* __restrict__ out, int Npad, int Kpad)
{
  size_t i = (size_t)blockIdx.x * blockDim.x + threadIdx.x;
  if (i >= (size_t)Npad * Kpad) return;
  int n = (int)(i / Kpad), k = (int)(i % Kpad);
  out[i] = (n < N && k < Kw) ? (__bf16)W[(size_t)n * Kw + k] : (__bf16)0.0f;
}

__global__ void f32_to_bf16(const float* __restrict__ in, __bf16* __restrict__ out, int n)
{
  int i = blockIdx.x * blockDim.x + threadIdx.x;
  if (i < n) out[i] = (__bf16)in[i];
}

__global__ void zero_f32(float* __restrict__ p, int n)
{
  int i = blockIdx.x * blockDim.x + threadIdx.x;
  if (i < n) p[i] = 0.0f;
}

// ---------------------------------------------------------------------------
// Guard-free bf16 WMMA GEMM: C[M,N] = act(A[M,lda f32] @ Wb[Npad,Kpad bf16]^T + bias)
// One wave computes a 16x64 tile (4 N-tiles sharing the A fragment).
// Kpad multiple of 32 with an EVEN number of 32-chunks; Npad multiple of 64;
// A must be zero-padded to Kpad. Ping-pong double buffering (no rotate movs).
// ---------------------------------------------------------------------------
__device__ __forceinline__ void load_a_frag(const float* __restrict__ arow,
                                            int k0, int half, v16bf& a)
{
  float af[16];
  *(float4*)&af[0]  = *(const float4*)(arow + k0 + half * 16);
  *(float4*)&af[4]  = *(const float4*)(arow + k0 + half * 16 + 4);
  *(float4*)&af[8]  = *(const float4*)(arow + k0 + half * 16 + 8);
  *(float4*)&af[12] = *(const float4*)(arow + k0 + half * 16 + 12);
#pragma unroll
  for (int j = 0; j < 16; ++j) a[j] = (__bf16)af[j];
}

__global__ __launch_bounds__(256) void gemm_bias_wmma(
    const float*  __restrict__ A, int lda,
    const __bf16* __restrict__ Wb, int Kpad,
    const float*  __restrict__ bias,
    float*        __restrict__ C, int ldc, int coff,
    int M, int N, int Npad, int relu)
{
  int wid  = blockIdx.x * (blockDim.x >> 5) + (threadIdx.x >> 5);
  int lane = threadIdx.x & 31;
  int tn  = Npad >> 6;              // 64-wide wave tiles
  int tm  = M >> 4;
  if (wid >= tm * tn) return;
  int mt = wid / tn, nt = wid % tn;
  int m0 = mt << 4, n0 = nt << 6;
  int half = lane >> 4, qr = lane & 15;

  const float* arow = A + (size_t)(m0 + qr) * lda;
  const __bf16* wrow[4];
#pragma unroll
  for (int t = 0; t < 4; ++t)
    wrow[t] = Wb + (size_t)(n0 + t * 16 + qr) * Kpad + half * 16;

  v16bf a0, a1, b0[4], b1[4];
  v8f   c[4] = {};

  // preload chunk 0 into buffer 0
  load_a_frag(arow, 0, half, a0);
#pragma unroll
  for (int t = 0; t < 4; ++t) b0[t] = *(const v16bf*)(wrow[t]);

  // main loop: process chunk pairs with ping-pong buffers (no rotation copies)
  int k = 0;
  for (; k + 64 < Kpad; k += 64) {
    load_a_frag(arow, k + 32, half, a1);
#pragma unroll
    for (int t = 0; t < 4; ++t) b1[t] = *(const v16bf*)(wrow[t] + k + 32);
    if (k + 160 < Kpad) __builtin_prefetch(arow + k + 160, 0, 0);  // global_prefetch_b8
#pragma unroll
    for (int t = 0; t < 4; ++t)
      c[t] = __builtin_amdgcn_wmma_f32_16x16x32_bf16(false, a0, false, b0[t],
                                                     (short)0, c[t], false, false);

    load_a_frag(arow, k + 64, half, a0);
#pragma unroll
    for (int t = 0; t < 4; ++t) b0[t] = *(const v16bf*)(wrow[t] + k + 64);
#pragma unroll
    for (int t = 0; t < 4; ++t)
      c[t] = __builtin_amdgcn_wmma_f32_16x16x32_bf16(false, a1, false, b1[t],
                                                     (short)0, c[t], false, false);
  }

  // peeled tail: last two chunks (chunk count is always even)
  load_a_frag(arow, k + 32, half, a1);
#pragma unroll
  for (int t = 0; t < 4; ++t) b1[t] = *(const v16bf*)(wrow[t] + k + 32);
#pragma unroll
  for (int t = 0; t < 4; ++t)
    c[t] = __builtin_amdgcn_wmma_f32_16x16x32_bf16(false, a0, false, b0[t],
                                                   (short)0, c[t], false, false);
#pragma unroll
  for (int t = 0; t < 4; ++t)
    c[t] = __builtin_amdgcn_wmma_f32_16x16x32_bf16(false, a1, false, b1[t],
                                                   (short)0, c[t], false, false);

#pragma unroll
  for (int t = 0; t < 4; ++t) {
    int n = n0 + t * 16 + qr;
    if (n < N) {
      float bv = bias ? bias[n] : 0.0f;
#pragma unroll
      for (int r = 0; r < 8; ++r) {          // VGPR r -> M = half*8 + r
        float v = c[t][r] + bv;
        if (relu) v = v > 0.0f ? v : 0.0f;
        C[(size_t)(m0 + half * 8 + r) * ldc + coff + n] = v;
      }
    }
  }
}

// ---------------------------------------------------------------------------
// Proprioceptive embed: ft[:,512+c] = relu(propri @ pW^T + pb), K=107 (tiny)
// ---------------------------------------------------------------------------
__global__ void propri_embed(const float* __restrict__ propri,
                             const float* __restrict__ pW,
                             const float* __restrict__ pb,
                             float* __restrict__ ft)
{
  int r = blockIdx.x;
  int c = threadIdx.x;
  const float* x = propri + (size_t)r * 107;
  const float* w = pW + (size_t)c * 107;
  float acc = pb[c];
  for (int k = 0; k < 107; ++k) acc = fmaf(x[k], w[k], acc);
  ft[(size_t)r * 640 + 512 + c] = acc > 0.0f ? acc : 0.0f;
}

// ---------------------------------------------------------------------------
// Core LSTM scan: hid=256, gates=1024. One persistent workgroup, 16 waves.
// h kept as bf16 in LDS (A fragments), Whh streamed as bf16 (L2-resident),
// gates staged through LDS, barrier per step.
// ---------------------------------------------------------------------------
__global__ __launch_bounds__(512) void core_lstm_scan(
    const float*  __restrict__ xW,    // [256*16, 1024]
    const __bf16* __restrict__ Whhb,  // [1024, 256] bf16
    const float*  __restrict__ bhh,   // [1024]
    const int*    __restrict__ done,  // [16, 256]
    float*        __restrict__ hs)    // [256*16, 256]
{
  __shared__ __bf16 hbf[16][256];
  __shared__ float  cst[16][256];
  __shared__ float  gates[16][1024];

  int tid = threadIdx.x;
  int lane = tid & 31, wv = tid >> 5;
  int half = lane >> 4, qr = lane & 15;

  for (int i = tid; i < 16 * 256; i += 512) {
    ((__bf16*)hbf)[i] = (__bf16)0.0f;
    ((float*)cst)[i] = 0.0f;
  }
  __syncthreads();

  for (int t = 0; t < 256; ++t) {
    if (t > 0) {
      for (int i = tid; i < 16 * 256; i += 512) {
        int b = i >> 8;
        if (done[b * 256 + (t - 1)] != 0) {
          ((__bf16*)hbf)[i] = (__bf16)0.0f;
          ((float*)cst)[i] = 0.0f;
        }
      }
      __syncthreads();
    }

#pragma unroll
    for (int ti = 0; ti < 4; ++ti) {
      int n0 = (wv * 4 + ti) << 4;
      const __bf16* wrow = Whhb + (size_t)(n0 + qr) * 256 + half * 16;
      v8f c = {};
#pragma unroll
      for (int kc = 0; kc < 8; ++kc) {
        int k0 = kc * 32;
        v8bf a0 = *(const v8bf*)&hbf[qr][k0 + half * 16];
        v8bf a1 = *(const v8bf*)&hbf[qr][k0 + half * 16 + 8];
        v16bf a, b;
#pragma unroll
        for (int j = 0; j < 8; ++j) { a[j] = a0[j]; a[8 + j] = a1[j]; }
        b = *(const v16bf*)(wrow + k0);
        c = __builtin_amdgcn_wmma_f32_16x16x32_bf16(false, a, false, b,
                                                    (short)0, c, false, false);
      }
#pragma unroll
      for (int r = 0; r < 8; ++r)
        gates[half * 8 + r][n0 + qr] = c[r];
    }
    __syncthreads();

    const float* xwt = xW + (size_t)t * 16 * 1024;
    for (int i = tid; i < 16 * 256; i += 512) {
      int b = i >> 8, n = i & 255;
      float gi = gates[b][n]       + xwt[b * 1024 + n]       + bhh[n];
      float gf = gates[b][256 + n] + xwt[b * 1024 + 256 + n] + bhh[256 + n];
      float gg = gates[b][512 + n] + xwt[b * 1024 + 512 + n] + bhh[512 + n];
      float go = gates[b][768 + n] + xwt[b * 1024 + 768 + n] + bhh[768 + n];
      float si = 1.0f / (1.0f + __expf(-gi));
      float sf = 1.0f / (1.0f + __expf(-gf));
      float so = 1.0f / (1.0f + __expf(-go));
      float cn = sf * cst[b][n] + si * tanhf(gg);
      float hn = so * tanhf(cn);
      cst[b][n] = cn;
      hbf[b][n] = (__bf16)hn;
      hs[(size_t)(t * 16 + b) * 256 + n] = hn;
    }
    __syncthreads();
  }
}

// ---------------------------------------------------------------------------
// policy_in = concat(core_out[256], ft[640], propri[107]) zero-padded to 1024
// ---------------------------------------------------------------------------
__global__ void concat_policy_in(const float* __restrict__ hs,
                                 const float* __restrict__ ft,
                                 const float* __restrict__ propri,
                                 float* __restrict__ out)
{
  size_t i = (size_t)blockIdx.x * blockDim.x + threadIdx.x;
  int r = (int)(i >> 10), c = (int)(i & 1023);
  float v;
  if (c < 256)       v = hs[(size_t)r * 256 + c];
  else if (c < 896)  v = ft[(size_t)r * 640 + (c - 256)];
  else if (c < 1003) v = propri[(size_t)r * 107 + (c - 896)];
  else               v = 0.0f;
  out[i] = v;
}

// ---------------------------------------------------------------------------
// Policy LSTM scan: hid=38, gates=152; tiny -> plain VALU, Whh in LDS.
// ---------------------------------------------------------------------------
__global__ __launch_bounds__(512) void policy_lstm_scan(
    const float* __restrict__ xW,    // [4096, 152]
    const float* __restrict__ Whh,   // [152, 38]
    const float* __restrict__ bhh,   // [152]
    const int*   __restrict__ done,
    float*       __restrict__ out, int ldo)
{
  __shared__ float wsh[152 * 38];
  __shared__ float h[16][38];
  __shared__ float cs[16][38];
  __shared__ float g[16][152];
  int tid = threadIdx.x;
  for (int i = tid; i < 152 * 38; i += 512) wsh[i] = Whh[i];
  for (int i = tid; i < 16 * 38; i += 512) { ((float*)h)[i] = 0.0f; ((float*)cs)[i] = 0.0f; }
  __syncthreads();

  for (int t = 0; t < 256; ++t) {
    if (t > 0) {
      for (int i = tid; i < 16 * 38; i += 512) {
        int b = i / 38;
        if (done[b * 256 + (t - 1)] != 0) { ((float*)h)[i] = 0.0f; ((float*)cs)[i] = 0.0f; }
      }
      __syncthreads();
    }
    const float* xwt = xW + (size_t)t * 16 * 152;
    for (int i = tid; i < 16 * 152; i += 512) {
      int b = i / 152, j = i % 152;
      float acc = xwt[b * 152 + j] + bhh[j];
      const float* wr = wsh + j * 38;
#pragma unroll
      for (int k = 0; k < 38; ++k) acc = fmaf(h[b][k], wr[k], acc);
      g[b][j] = acc;
    }
    __syncthreads();
    for (int i = tid; i < 16 * 38; i += 512) {
      int b = i / 38, n = i % 38;
      float si = 1.0f / (1.0f + __expf(-g[b][n]));
      float sf = 1.0f / (1.0f + __expf(-g[b][38 + n]));
      float gg = tanhf(g[b][76 + n]);
      float so = 1.0f / (1.0f + __expf(-g[b][114 + n]));
      float cn = sf * cs[b][n] + si * gg;
      float hn = so * tanhf(cn);
      cs[b][n] = cn; h[b][n] = hn;
      out[(size_t)(t * 16 + b) * ldo + n] = hn;
    }
    __syncthreads();
  }
}

// ---------------------------------------------------------------------------
__global__ void value_out(const float* __restrict__ v,
                          const float* __restrict__ W,
                          const float* __restrict__ b,
                          float* __restrict__ out)
{
  int r = blockIdx.x * blockDim.x + threadIdx.x;
  if (r >= 4096) return;
  const float* x = v + (size_t)r * 256;
  float acc = b[0];
  for (int k = 0; k < 256; ++k) acc = fmaf(x[k], W[k], acc);
  out[r] = acc;
}

// ---------------------------------------------------------------------------
extern "C" void kernel_launch(void* const* d_in, const int* in_sizes, int n_in,
                              void* d_out, int out_size, void* d_ws, size_t ws_size,
                              hipStream_t stream)
{
  const float* visual  = (const float*)d_in[0];
  const float* propri  = (const float*)d_in[1];
  const int*   done    = (const int*)d_in[2];
  const float* vW      = (const float*)d_in[3];
  const float* vb      = (const float*)d_in[4];
  const float* pW      = (const float*)d_in[5];
  const float* pb      = (const float*)d_in[6];
  const float* cWih    = (const float*)d_in[7];
  const float* cWhh    = (const float*)d_in[8];
  const float* cbih    = (const float*)d_in[9];
  const float* cbhh    = (const float*)d_in[10];
  const float* pWih[3] = {(const float*)d_in[11], (const float*)d_in[15], (const float*)d_in[19]};
  const float* pWhh[3] = {(const float*)d_in[12], (const float*)d_in[16], (const float*)d_in[20]};
  const float* pbih[3] = {(const float*)d_in[13], (const float*)d_in[17], (const float*)d_in[21]};
  const float* pbhh[3] = {(const float*)d_in[14], (const float*)d_in[18], (const float*)d_in[22]};
  const float* vhW[5]  = {(const float*)d_in[23], (const float*)d_in[25], (const float*)d_in[27],
                          (const float*)d_in[29], (const float*)d_in[31]};
  const float* vhB[5]  = {(const float*)d_in[24], (const float*)d_in[26], (const float*)d_in[28],
                          (const float*)d_in[30], (const float*)d_in[32]};
  const float* valW    = (const float*)d_in[33];
  const float* valB    = (const float*)d_in[34];

  float* out = (float*)d_out;

  // ---- workspace layout (bytes) ----
  char* ws = (char*)d_ws;
  float*  ft    = (float*) (ws + 0);          // [4096,640]
  float*  xWc   = (float*) (ws + 10485760);   // [4096,1024]
  float*  hs    = (float*) (ws + 27262976);   // [4096,256]
  float*  pin   = (float*) (ws + 31457280);   // [4096,1024]
  float*  xWp   = (float*) (ws + 48234496);   // [4096,152]
  float*  hp1   = (float*) (ws + 50724864);   // [4096,64] zero-padded
  float*  hp2   = (float*) (ws + 51773440);   // [4096,64] zero-padded
  float*  vb0   = (float*) (ws + 52822016);   // [4096,256]
  float*  vb1   = (float*) (ws + 57016320);   // [4096,256]
  __bf16* WhhB  = (__bf16*)(ws + 61210624);   // [1024,256]
  __bf16* vWb   = (__bf16*)(ws + 61734912);   // [512,12288]
  __bf16* cWihb = (__bf16*)(ws + 74317824);   // [1024,640]
  __bf16* pW1b  = (__bf16*)(ws + 75628544);   // [192,1024]
  __bf16* pW2b  = (__bf16*)(ws + 76021760);   // [192,64]
  __bf16* pW3b  = (__bf16*)(ws + 76046336);   // [192,64]
  __bf16* vhWb0 = (__bf16*)(ws + 76070912);   // 5 x [256,256]

  (void)in_sizes; (void)n_in; (void)out_size; (void)ws_size;

  // ---- weight packing (zero-padded bf16) ----
  f32_to_bf16<<<1024, 256, 0, stream>>>(cWhh, WhhB, 1024 * 256);
  pack_w_bf16<<<24576, 256, 0, stream>>>(vW, 512, 12288, vWb, 512, 12288);
  pack_w_bf16<<<2560, 256, 0, stream>>>(cWih, 1024, 640, cWihb, 1024, 640);
  pack_w_bf16<<<768, 256, 0, stream>>>(pWih[0], 152, 1003, pW1b, 192, 1024);
  pack_w_bf16<<<48, 256, 0, stream>>>(pWih[1], 152, 38, pW2b, 192, 64);
  pack_w_bf16<<<48, 256, 0, stream>>>(pWih[2], 152, 38, pW3b, 192, 64);
  for (int l = 0; l < 5; ++l)
    pack_w_bf16<<<256, 256, 0, stream>>>(vhW[l], 256, 256, vhWb0 + (size_t)l * 65536, 256, 256);
  zero_f32<<<1024, 256, 0, stream>>>(hp1, 4096 * 64);
  zero_f32<<<1024, 256, 0, stream>>>(hp2, 4096 * 64);

  // 1) visual embed: ft[:, :512] = relu(visual @ vW^T + vb)  (2048 waves)
  gemm_bias_wmma<<<256, 256, 0, stream>>>(visual, 12288, vWb, 12288,
                                          vb, ft, 640, 0, 4096, 512, 512, 1);
  // 2) propri embed: ft[:, 512:640]
  propri_embed<<<4096, 128, 0, stream>>>(propri, pW, pb, ft);

  // 3) core xW = ft @ Wih^T + bih : [4096,1024]  (4096 waves)
  gemm_bias_wmma<<<512, 256, 0, stream>>>(ft, 640, cWihb, 640,
                                          cbih, xWc, 1024, 0, 4096, 1024, 1024, 0);

  // 4) core LSTM scan (persistent workgroup, WMMA recurrent)
  core_lstm_scan<<<1, 512, 0, stream>>>(xWc, WhhB, cbhh, done, hs);

  // 5) policy_in = concat(hs, ft, propri) padded to 1024
  concat_policy_in<<<16384, 256, 0, stream>>>(hs, ft, propri, pin);

  // 6) policy layer 1
  gemm_bias_wmma<<<96, 256, 0, stream>>>(pin, 1024, pW1b, 1024,
                                         pbih[0], xWp, 152, 0, 4096, 152, 192, 0);
  policy_lstm_scan<<<1, 512, 0, stream>>>(xWp, pWhh[0], pbhh[0], done, hp1, 64);

  // 7) policy layer 2
  gemm_bias_wmma<<<96, 256, 0, stream>>>(hp1, 64, pW2b, 64,
                                         pbih[1], xWp, 152, 0, 4096, 152, 192, 0);
  policy_lstm_scan<<<1, 512, 0, stream>>>(xWp, pWhh[1], pbhh[1], done, hp2, 64);

  // 8) policy layer 3 -> policy output directly into d_out[4096..]
  gemm_bias_wmma<<<96, 256, 0, stream>>>(hp2, 64, pW3b, 64,
                                         pbih[2], xWp, 152, 0, 4096, 152, 192, 0);
  policy_lstm_scan<<<1, 512, 0, stream>>>(xWp, pWhh[2], pbhh[2], done, out + 4096, 38);

  // 9) value MLP: 5 x relu(256x256), then final dot -> d_out[0..4095]
  const float* vin = hs;
  float* vpp[2] = {vb0, vb1};
  for (int l = 0; l < 5; ++l) {
    float* vout = vpp[l & 1];
    gemm_bias_wmma<<<128, 256, 0, stream>>>(vin, 256, vhWb0 + (size_t)l * 65536, 256,
                                            vhB[l], vout, 256, 0, 4096, 256, 256, 1);
    vin = vout;
  }
  value_out<<<16, 256, 0, stream>>>(vin, valW, valB, out);
}
